// LongformerSelfAttentionForBart_70282844832208
// MI455X (gfx1250) — matrix operand order repaired and checked by reading, not compiled, not run on hardware
//
#include <hip/hip_runtime.h>

typedef unsigned short u16;
typedef unsigned int u32;
typedef __attribute__((ext_vector_type(16))) __bf16 v16bf;
typedef __attribute__((ext_vector_type(8)))  float  v8f;

#define NEGV (-1.0e9f)

union FragAB { v16bf v; u32 u[8]; };

__device__ __forceinline__ u16 f2bf(float f) {
    u32 u = __float_as_uint(f);
    return (u16)((u + 0x7FFFu + ((u >> 16) & 1u)) >> 16);
}

__device__ __forceinline__ v8f wmma_bf16(v16bf a, v16bf b, v8f c) {
    return __builtin_amdgcn_wmma_f32_16x16x32_bf16(false, a, false, b, (short)0, c, false, false);
}

// A fragment (16x32 bf16): element (m,k) = base[(m0+m)*stride + kbase + k]
__device__ __forceinline__ v16bf load_a(const u16* __restrict__ base, int stride, int m0, int kbase) {
    int lane = threadIdx.x & 31;
    int M = lane & 15, half = lane >> 4;
    const u16* p = base + (size_t)(m0 + M) * stride + kbase + 8 * half;
    FragAB f;
#pragma unroll
    for (int v = 0; v < 4; ++v) f.u[v] = *(const u32*)(p + 2 * v);
#pragma unroll
    for (int v = 4; v < 8; ++v) f.u[v] = *(const u32*)(p + 2 * v + 8);
    return f.v;
}

// B fragment (32x16 bf16), B[k][n] = base[(n0+n)*stride + kbase + k]
__device__ __forceinline__ v16bf load_b(const u16* __restrict__ base, int stride, int n0, int kbase) {
    int lane = threadIdx.x & 31;
    int N = lane & 15, half = lane >> 4;
    const u16* p = base + (size_t)(n0 + N) * stride + kbase + 8 * half;
    FragAB f;
#pragma unroll
    for (int v = 0; v < 4; ++v) f.u[v] = *(const u32*)(p + 2 * v);
#pragma unroll
    for (int v = 4; v < 8; ++v) f.u[v] = *(const u32*)(p + 2 * v + 8);
    return f.v;
}

// ---- convert kernels -------------------------------------------------------
// query (S,B,D) f32 -> X (B,S,D) bf16
__global__ void __launch_bounds__(256, 1) cvt_x(const float* __restrict__ q, u16* __restrict__ x) {
    int o = blockIdx.x * 256 + threadIdx.x;          // < 6291456
    int d = o % 768;
    int sb = o / 768;                                 // (b*4096 + s)
    int b = sb >> 12, s = sb & 4095;
    x[o] = f2bf(q[((size_t)s * 2 + b) * 768 + d]);
}

// W (K,N) f32 -> Wt (N,K) bf16
__global__ void __launch_bounds__(256, 1) cvt_wt(const float* __restrict__ w, u16* __restrict__ wt) {
    int o = blockIdx.x * 256 + threadIdx.x;          // < 589824
    int k = o % 768, n = o / 768;
    wt[o] = f2bf(w[(size_t)k * 768 + n]);
}

// ---- projection / output GEMM ---------------------------------------------
// C(8192x768) = A(8192x768) @ Wt^T + bias, optional scale. 32x64 tile per wave.
// MODE 0: bf16 out (B,H,S,HD)   MODE 1: bf16 out (B,H,HD,S)   MODE 2: f32 out (S,B,D)
template <int MODE>
__global__ void __launch_bounds__(256, 1)
gemm768(const u16* __restrict__ A, const u16* __restrict__ Wt,
        const float* __restrict__ bias, float scale, void* __restrict__ outp) {
    int waveId = blockIdx.x * 8 + (threadIdx.x >> 5);
    int tm = waveId / 12, tn = waveId % 12;
    int m0 = tm * 32, n0 = tn * 64;
    int lane = threadIdx.x & 31, half = lane >> 4;

    v8f z = {0.f, 0.f, 0.f, 0.f, 0.f, 0.f, 0.f, 0.f};
    v8f c[2][4] = {{z, z, z, z}, {z, z, z, z}};

    for (int kk = 0; kk < 24; ++kk) {
        int kb = kk * 32;
        if (kk + 2 < 24) {   // cover the load->wmma latency chain
            __builtin_prefetch(A + (size_t)(m0 + (lane & 15)) * 768 + kb + 64, 0, 1);
            __builtin_prefetch(Wt + (size_t)(n0 + (lane & 31)) * 768 + kb + 64, 0, 1);
        }
        v16bf a0 = load_a(A, 768, m0, kb);
        v16bf a1 = load_a(A, 768, m0 + 16, kb);
#pragma unroll
        for (int j = 0; j < 4; ++j) {
            v16bf bj = load_b(Wt, 768, n0 + 16 * j, kb);
            c[0][j] = wmma_bf16(a0, bj, c[0][j]);
            c[1][j] = wmma_bf16(a1, bj, c[1][j]);
        }
    }
#pragma unroll
    for (int i = 0; i < 2; ++i) {
#pragma unroll
        for (int j = 0; j < 4; ++j) {
            int col = n0 + j * 16 + (lane & 15);
            float bv = bias[col];
#pragma unroll
            for (int v = 0; v < 8; ++v) {
                int m = m0 + i * 16 + v + 8 * half;
                float val = (c[i][j][v] + bv) * scale;
                int b = m >> 12, s = m & 4095;
                int h = col >> 6, d = col & 63;
                if (MODE == 0) {
                    ((u16*)outp)[(((size_t)(b * 12 + h) * 4096 + s) << 6) + d] = f2bf(val);
                } else if (MODE == 1) {
                    ((u16*)outp)[(((size_t)(b * 12 + h) * 64 + d) << 12) + s] = f2bf(val);
                } else {
                    ((float*)outp)[((size_t)s * 2 + b) * 768 + col] = val;
                }
            }
        }
    }
}

// ---- local (sliding window + global-key) attention -------------------------
// grid: B*H*(S/64) blocks of 128 threads; each wave owns 16 query rows.
__global__ void __launch_bounds__(128, 1)
local_attn(const u16* __restrict__ Qb, const u16* __restrict__ Kb,
           const u16* __restrict__ Vt, const int* __restrict__ kpm,
           u16* __restrict__ AO) {
    __shared__ __align__(16) u16 Plds[4 * 16 * 64];
    int bh = blockIdx.x >> 6;
    int chunk = blockIdx.x & 63;
    int b = bh / 12, h = bh % 12;
    int wave = threadIdx.x >> 5, lane = threadIdx.x & 31, half = lane >> 4;
    int i0b = chunk * 64;
    int i0 = i0b + wave * 16;
    u16* P = Plds + wave * (16 * 64);

    const u16* Qp = Qb + (size_t)bh * 4096 * 64;
    const u16* Kp = Kb + (size_t)bh * 4096 * 64;
    const u16* Vp = Vt + (size_t)bh * 64 * 4096;
    const int* mrow = kpm + b * 4096;

    v16bf aq0 = load_a(Qp, 64, i0, 0);
    v16bf aq1 = load_a(Qp, 64, i0, 32);

    v8f z = {0.f, 0.f, 0.f, 0.f, 0.f, 0.f, 0.f, 0.f};
    v8f o[4] = {z, z, z, z};
    float rm[8], l[8];
#pragma unroll
    for (int v = 0; v < 8; ++v) { rm[v] = -1e30f; l[v] = 0.f; }

    // block-uniform local key-block range covering rows [i0b, i0b+63]
    int lo = i0b - 256; if (lo < 0) lo = 0;
    int hi = i0b + 63 + 256; if (hi > 4095) hi = 4095;
    int jlo = lo & ~63, jhi = hi & ~63;
    int nloc = ((jhi - jlo) >> 6) + 1;

    for (int t = 0; t < 1 + nloc; ++t) {
        bool isg = (t == 0);
        int j0 = isg ? 0 : jlo + ((t - 1) << 6);

        v8f s[4];
#pragma unroll
        for (int n = 0; n < 4; ++n) {
            s[n] = z;
            s[n] = wmma_bf16(aq0, load_b(Kp, 64, j0 + n * 16, 0), s[n]);
            s[n] = wmma_bf16(aq1, load_b(Kp, 64, j0 + n * 16, 32), s[n]);
        }
        if (!isg) {
            int cb = j0 + (lane & 15);
#pragma unroll
            for (int n = 0; n < 4; ++n) {
                int j = cb + n * 16;
                int bad = mrow[j];
#pragma unroll
                for (int v = 0; v < 8; ++v) {
                    int i = i0 + v + 8 * half;
                    if (bad != 0 || j < i - 256 || j > i + 256) s[n][v] = NEGV;
                }
            }
        }
        // online softmax update
        float sc[8], rs[8];
#pragma unroll
        for (int v = 0; v < 8; ++v) {
            float m4 = fmaxf(fmaxf(s[0][v], s[1][v]), fmaxf(s[2][v], s[3][v]));
            m4 = fmaxf(m4, __shfl_xor(m4, 8, 32));
            m4 = fmaxf(m4, __shfl_xor(m4, 4, 32));
            m4 = fmaxf(m4, __shfl_xor(m4, 2, 32));
            m4 = fmaxf(m4, __shfl_xor(m4, 1, 32));
            float nm = fmaxf(rm[v], m4);
            sc[v] = __expf(rm[v] - nm);
            rm[v] = nm;
            rs[v] = 0.f;
        }
#pragma unroll
        for (int n = 0; n < 4; ++n)
#pragma unroll
            for (int v = 0; v < 8; ++v) {
                float p = __expf(s[n][v] - rm[v]);
                s[n][v] = p;
                rs[v] += p;
            }
#pragma unroll
        for (int v = 0; v < 8; ++v) {
            float r = rs[v];
            r += __shfl_xor(r, 8, 32);
            r += __shfl_xor(r, 4, 32);
            r += __shfl_xor(r, 2, 32);
            r += __shfl_xor(r, 1, 32);
            l[v] = l[v] * sc[v] + r;
        }
#pragma unroll
        for (int n = 0; n < 4; ++n)
#pragma unroll
            for (int v = 0; v < 8; ++v) o[n][v] *= sc[v];

        // stage P (C layout -> bf16) in LDS, reload as A fragments
#pragma unroll
        for (int n = 0; n < 4; ++n)
#pragma unroll
            for (int v = 0; v < 8; ++v)
                P[(v + 8 * half) * 64 + n * 16 + (lane & 15)] = f2bf(s[n][v]);
        __syncthreads();
        v16bf ap0 = load_a(P, 64, 0, 0);
        v16bf ap1 = load_a(P, 64, 0, 32);
#pragma unroll
        for (int n = 0; n < 4; ++n) {
            o[n] = wmma_bf16(ap0, load_b(Vp, 4096, n * 16, j0), o[n]);
            o[n] = wmma_bf16(ap1, load_b(Vp, 4096, n * 16, j0 + 32), o[n]);
        }
        __syncthreads();
    }

#pragma unroll
    for (int n = 0; n < 4; ++n) {
        int d = n * 16 + (lane & 15);
#pragma unroll
        for (int v = 0; v < 8; ++v) {
            int i = i0 + v + 8 * half;
            AO[((size_t)(b * 4096 + i)) * 768 + h * 64 + d] = f2bf(o[n][v] / l[v]);
        }
    }
}

// ---- global attention for the first G=64 rows ------------------------------
// grid: B*H blocks of 128 threads; wave owns 16 of the 64 global query rows.
__global__ void __launch_bounds__(128, 1)
global_attn(const u16* __restrict__ Qg, const u16* __restrict__ Kg,
            const u16* __restrict__ Vgt, const int* __restrict__ kpm,
            u16* __restrict__ AO) {
    __shared__ __align__(16) u16 Plds[4 * 16 * 64];
    int bh = blockIdx.x;
    int b = bh / 12, h = bh % 12;
    int wave = threadIdx.x >> 5, lane = threadIdx.x & 31, half = lane >> 4;
    int g0 = wave * 16;
    u16* P = Plds + wave * (16 * 64);

    const u16* Qp = Qg + (size_t)bh * 4096 * 64;
    const u16* Kp = Kg + (size_t)bh * 4096 * 64;
    const u16* Vp = Vgt + (size_t)bh * 64 * 4096;
    const int* mrow = kpm + b * 4096;

    v16bf aq0 = load_a(Qp, 64, g0, 0);
    v16bf aq1 = load_a(Qp, 64, g0, 32);

    v8f z = {0.f, 0.f, 0.f, 0.f, 0.f, 0.f, 0.f, 0.f};
    v8f o[4] = {z, z, z, z};
    float rm[8], l[8];
#pragma unroll
    for (int v = 0; v < 8; ++v) { rm[v] = -1e30f; l[v] = 0.f; }

    for (int j0 = 0; j0 < 4096; j0 += 64) {
        v8f s[4];
#pragma unroll
        for (int n = 0; n < 4; ++n) {
            s[n] = z;
            s[n] = wmma_bf16(aq0, load_b(Kp, 64, j0 + n * 16, 0), s[n]);
            s[n] = wmma_bf16(aq1, load_b(Kp, 64, j0 + n * 16, 32), s[n]);
        }
#pragma unroll
        for (int n = 0; n < 4; ++n) {
            int j = j0 + n * 16 + (lane & 15);
            if (mrow[j] > 0) {
#pragma unroll
                for (int v = 0; v < 8; ++v) s[n][v] = NEGV;
            }
        }
        float sc[8], rs[8];
#pragma unroll
        for (int v = 0; v < 8; ++v) {
            float m4 = fmaxf(fmaxf(s[0][v], s[1][v]), fmaxf(s[2][v], s[3][v]));
            m4 = fmaxf(m4, __shfl_xor(m4, 8, 32));
            m4 = fmaxf(m4, __shfl_xor(m4, 4, 32));
            m4 = fmaxf(m4, __shfl_xor(m4, 2, 32));
            m4 = fmaxf(m4, __shfl_xor(m4, 1, 32));
            float nm = fmaxf(rm[v], m4);
            sc[v] = __expf(rm[v] - nm);
            rm[v] = nm;
            rs[v] = 0.f;
        }
#pragma unroll
        for (int n = 0; n < 4; ++n)
#pragma unroll
            for (int v = 0; v < 8; ++v) {
                float p = __expf(s[n][v] - rm[v]);
                s[n][v] = p;
                rs[v] += p;
            }
#pragma unroll
        for (int v = 0; v < 8; ++v) {
            float r = rs[v];
            r += __shfl_xor(r, 8, 32);
            r += __shfl_xor(r, 4, 32);
            r += __shfl_xor(r, 2, 32);
            r += __shfl_xor(r, 1, 32);
            l[v] = l[v] * sc[v] + r;
        }
#pragma unroll
        for (int n = 0; n < 4; ++n)
#pragma unroll
            for (int v = 0; v < 8; ++v) o[n][v] *= sc[v];

#pragma unroll
        for (int n = 0; n < 4; ++n)
#pragma unroll
            for (int v = 0; v < 8; ++v)
                P[(v + 8 * half) * 64 + n * 16 + (lane & 15)] = f2bf(s[n][v]);
        __syncthreads();
        v16bf ap0 = load_a(P, 64, 0, 0);
        v16bf ap1 = load_a(P, 64, 0, 32);
#pragma unroll
        for (int n = 0; n < 4; ++n) {
            o[n] = wmma_bf16(ap0, load_b(Vp, 4096, n * 16, j0), o[n]);
            o[n] = wmma_bf16(ap1, load_b(Vp, 4096, n * 16, j0 + 32), o[n]);
        }
        __syncthreads();
    }

#pragma unroll
    for (int n = 0; n < 4; ++n) {
        int d = n * 16 + (lane & 15);
#pragma unroll
        for (int v = 0; v < 8; ++v) {
            int i = g0 + v + 8 * half;   // < 64 = G
            AO[((size_t)(b * 4096 + i)) * 768 + h * 64 + d] = f2bf(o[n][v] / l[v]);
        }
    }
}

extern "C" void kernel_launch(void* const* d_in, const int* in_sizes, int n_in,
                              void* d_out, int out_size, void* d_ws, size_t ws_size,
                              hipStream_t stream) {
    (void)in_sizes; (void)n_in; (void)out_size; (void)ws_size;
    const float* q_in = (const float*)d_in[0];
    const int*   kpm  = (const int*)d_in[1];
    const float* Wsrc[7] = {(const float*)d_in[2],  (const float*)d_in[4],
                            (const float*)d_in[6],  (const float*)d_in[8],
                            (const float*)d_in[10], (const float*)d_in[12],
                            (const float*)d_in[14]};
    const float* bq  = (const float*)d_in[3];
    const float* bk  = (const float*)d_in[5];
    const float* bv  = (const float*)d_in[7];
    const float* bqg = (const float*)d_in[9];
    const float* bkg = (const float*)d_in[11];
    const float* bvg = (const float*)d_in[13];
    const float* bo  = (const float*)d_in[15];
    float* out = (float*)d_out;

    const size_t NBSD = (size_t)2 * 4096 * 768;   // 6,291,456 elements
    const size_t NW   = (size_t)768 * 768;        //   589,824 elements
    u16* ws  = (u16*)d_ws;
    u16* Xbf = ws;               ws += NBSD;
    u16* WT[7];
    for (int i = 0; i < 7; ++i) { WT[i] = ws; ws += NW; }
    u16* Qb  = ws; ws += NBSD;
    u16* Kb  = ws; ws += NBSD;
    u16* Vt  = ws; ws += NBSD;
    u16* Qgb = ws; ws += NBSD;
    u16* Kgb = ws; ws += NBSD;
    u16* Vgt = ws; ws += NBSD;
    u16* AO  = ws; ws += NBSD;

    cvt_x<<<(int)(NBSD / 256), 256, 0, stream>>>(q_in, Xbf);
    for (int i = 0; i < 7; ++i)
        cvt_wt<<<(int)(NW / 256), 256, 0, stream>>>(Wsrc[i], WT[i]);

    gemm768<0><<<384, 256, 0, stream>>>(Xbf, WT[0], bq,  0.125f, Qb);
    gemm768<0><<<384, 256, 0, stream>>>(Xbf, WT[1], bk,  1.0f,   Kb);
    gemm768<1><<<384, 256, 0, stream>>>(Xbf, WT[2], bv,  1.0f,   Vt);
    gemm768<0><<<384, 256, 0, stream>>>(Xbf, WT[3], bqg, 0.125f, Qgb);
    gemm768<0><<<384, 256, 0, stream>>>(Xbf, WT[4], bkg, 1.0f,   Kgb);
    gemm768<1><<<384, 256, 0, stream>>>(Xbf, WT[5], bvg, 1.0f,   Vgt);

    local_attn<<<24 * 64, 128, 0, stream>>>(Qb, Kb, Vt, kpm, AO);
    global_attn<<<24, 128, 0, stream>>>(Qgb, Kgb, Vgt, kpm, AO);

    gemm768<2><<<384, 256, 0, stream>>>(AO, WT[6], bo, 1.0f, out);
}
